// SA_dotatt_8057358647876
// MI455X (gfx1250) — matrix operand order repaired and checked by reading, not compiled, not run on hardware
//
#include <hip/hip_runtime.h>
#include <hip/hip_bf16.h>

typedef __attribute__((ext_vector_type(16))) __bf16 v16bf;
typedef __attribute__((ext_vector_type(8)))  __bf16 v8bf;
typedef __attribute__((ext_vector_type(8)))  float  v8f;

#define BDIM 4
#define CDIM 128
#define NPIX 4096      // 64*64 pixels
#define KPROJ 1024     // C*L
#define NT   (NPIX / 64)

#define KPAD 40        // padded K-stride (halves) for 32-deep tiles
#define CPAD 136       // padded stride for 128-deep (c) rows
#define MPAD 72        // padded stride for 64-deep (m) rows

// ---- CDNA5 async global->LDS copy (ASYNCcnt-tracked) ----
static __device__ inline void async_ld16(unsigned lds_off, const __bf16* g) {
  asm volatile("global_load_async_to_lds_b128 %0, %1, off"
               :: "v"(lds_off), "v"(g) : "memory");
}
static __device__ inline void wait_async0() {
  asm volatile("s_wait_asynccnt 0" ::: "memory");
}
static __device__ inline void wait_async16() {   // <=16 outstanding: previous tile done
  asm volatile("s_wait_asynccnt 16" ::: "memory");
}
static __device__ inline unsigned lds_off32(const void* p) {
  return (unsigned)(unsigned long long)p;   // generic LDS ptr: low 32 bits = LDS offset
}

// ---- WMMA fragment helpers (bf16, contiguous k = 16*(lane>>4) + s in A and B) ----
static __device__ inline v16bf join16(v8bf lo, v8bf hi) {
  v16bf r;
#pragma unroll
  for (int i = 0; i < 8; ++i) { r[i] = lo[i]; r[i + 8] = hi[i]; }
  return r;
}

// A 16x32 (MxK): lane&15 = row M; halves = K kg..kg+15,
// kg = 16*(lane>>4). Two aligned b128 LDS loads.
static __device__ inline v16bf fragA(const __bf16* base, int lda, int lane) {
  const __bf16* p = base + (lane & 15) * lda + ((lane >> 4) << 4);
  return join16(*(const v8bf*)p, *(const v8bf*)(p + 8));
}

// B 32x16 (KxN) from TRANSPOSED tile BsT[n][k]: lane&15 = column N; halves =
// K kg..kg+15, kg = 16*(lane>>4). Two aligned b128 LDS loads.
static __device__ inline v16bf fragBT(const __bf16* baseT, int ldb, int lane) {
  const __bf16* p = baseT + (lane & 15) * ldb + ((lane >> 4) << 4);
  return join16(*(const v8bf*)p, *(const v8bf*)(p + 8));
}

static __device__ inline v8f wmma_bf16(v16bf a, v16bf b, v8f c) {
  return __builtin_amdgcn_wmma_f32_16x16x32_bf16(
      false, a, false, b, (short)0, c, false, false);
}

static __device__ inline unsigned pack_bf16(float a, float b) {
  union { unsigned u; __bf16 h[2]; } p;
  p.h[0] = (__bf16)a; p.h[1] = (__bf16)b;
  return p.u;
}

// ---- Kernel 1: projection GEMM. x[b,o,n] = sum_k proj_w[o,k] feats[b,k,n] + pb[o]
// Output stored TRANSPOSED: xT[b][n][c] (bf16) so downstream staging is contiguous.
__global__ __launch_bounds__(128) void proj_kernel(
    const float* __restrict__ feats, const float* __restrict__ proj_w,
    const float* __restrict__ proj_b, __bf16* __restrict__ xT) {
  __shared__ __bf16 As[64 * KPAD];     // A[o][k]
  __shared__ __bf16 BsT[64 * KPAD];    // B^T[n][k]
  unsigned* As32  = (unsigned*)As;
  unsigned* BsT32 = (unsigned*)BsT;
  const int tid  = threadIdx.x;
  const int lane = tid & 31;
  const int wv   = tid >> 5;
  const int nbase = blockIdx.x * 64;
  const int obase = blockIdx.y * 64;
  const int b     = blockIdx.z;

  v8f acc[4] = {};

  for (int kc = 0; kc < KPROJ / 32; ++kc) {
    const int kb = kc * 32;
#pragma unroll
    for (int t = 0; t < 8; ++t) {      // A tile: 64x32 f32 -> packed bf16 dwords
      int idx = t * 128 + tid;
      int o = idx >> 4, kp = idx & 15;
      const float2 f = *(const float2*)(proj_w + (size_t)(obase + o) * KPROJ + kb + 2 * kp);
      As32[o * (KPAD / 2) + kp] = pack_bf16(f.x, f.y);
    }
#pragma unroll
    for (int t = 0; t < 8; ++t) {      // B tile transposed: BsT[n][k]
      int idx = t * 128 + tid;
      int n = idx & 63, kp = idx >> 6;
      float f0 = feats[(size_t)(b * KPROJ + kb + 2 * kp) * NPIX + nbase + n];
      float f1 = feats[(size_t)(b * KPROJ + kb + 2 * kp + 1) * NPIX + nbase + n];
      BsT32[n * (KPAD / 2) + kp] = pack_bf16(f0, f1);
    }
    __syncthreads();
    v16bf a = fragA(As + wv * 16 * KPAD, KPAD, lane);
#pragma unroll
    for (int j = 0; j < 4; ++j) {
      v16bf bb = fragBT(BsT + j * 16 * KPAD, KPAD, lane);
      acc[j] = wmma_bf16(a, bb, acc[j]);
    }
    __syncthreads();
  }
  const int col = lane & 15;
  const int hi  = (lane >> 4) << 3;
#pragma unroll
  for (int j = 0; j < 4; ++j) {
#pragma unroll
    for (int r = 0; r < 8; ++r) {
      int o = obase + wv * 16 + r + hi;
      int n = nbase + j * 16 + col;
      xT[(size_t)(b * NPIX + n) * CDIM + o] = (__bf16)(acc[j][r] + proj_b[o]);
    }
  }
}

// ---- Kernel 2: q/k/v. grid.z = b*3 + which. q,k stored transposed [b][n][c];
// v stored [b][c][n]. x-tile staged via async global->LDS b128.
__global__ __launch_bounds__(128) void qkv_kernel(
    const __bf16* __restrict__ xT,
    const float* __restrict__ w1, const float* __restrict__ bias1,
    const float* __restrict__ w2, const float* __restrict__ bias2,
    const float* __restrict__ w3, const float* __restrict__ bias3,
    __bf16* __restrict__ qT, __bf16* __restrict__ kT, __bf16* __restrict__ vM) {
  __shared__ __bf16 As[64 * KPAD];
  __shared__ __bf16 BsT[64 * KPAD];
  unsigned* As32 = (unsigned*)As;
  const unsigned bs_o = lds_off32(BsT);
  const int tid  = threadIdx.x;
  const int lane = tid & 31;
  const int wv   = tid >> 5;
  const int nbase = blockIdx.x * 64;
  const int obase = blockIdx.y * 64;
  const int b     = blockIdx.z / 3;
  const int which = blockIdx.z % 3;
  const float* w    = (which == 0) ? w1 : (which == 1) ? w2 : w3;
  const float* bias = (which == 0) ? bias1 : (which == 1) ? bias2 : bias3;

  v8f acc[4] = {};

  for (int kc = 0; kc < CDIM / 32; ++kc) {
    const int kb = kc * 32;
    // B tile: xT rows are contiguous -> async 16B copies into padded BsT rows
#pragma unroll
    for (int t = 0; t < 2; ++t) {
      int idx = t * 128 + tid;
      int n = idx >> 2, seg = idx & 3;
      async_ld16(bs_o + n * (KPAD * 2) + seg * 16,
                 xT + (size_t)(b * NPIX + nbase + n) * CDIM + kb + seg * 8);
    }
#pragma unroll
    for (int t = 0; t < 8; ++t) {      // A tile: weights f32 -> bf16
      int idx = t * 128 + tid;
      int o = idx >> 4, kp = idx & 15;
      const float2 f = *(const float2*)(w + (size_t)(obase + o) * CDIM + kb + 2 * kp);
      As32[o * (KPAD / 2) + kp] = pack_bf16(f.x, f.y);
    }
    wait_async0();
    __syncthreads();
    v16bf a = fragA(As + wv * 16 * KPAD, KPAD, lane);
#pragma unroll
    for (int j = 0; j < 4; ++j) {
      v16bf bb = fragBT(BsT + j * 16 * KPAD, KPAD, lane);
      acc[j] = wmma_bf16(a, bb, acc[j]);
    }
    __syncthreads();
  }
  const int col = lane & 15;
  const int hi  = (lane >> 4) << 3;
#pragma unroll
  for (int j = 0; j < 4; ++j) {
#pragma unroll
    for (int r = 0; r < 8; ++r) {
      int o = obase + wv * 16 + r + hi;
      int n = nbase + j * 16 + col;
      __bf16 val = (__bf16)(acc[j][r] + bias[o]);
      if (which == 0)      qT[(size_t)(b * NPIX + n) * CDIM + o] = val;
      else if (which == 1) kT[(size_t)(b * NPIX + n) * CDIM + o] = val;
      else                 vM[(size_t)(b * CDIM + o) * NPIX + n] = val;
    }
  }
}

// ---- Kernel 3: fused flash attention with DOUBLE-BUFFERED async K/V staging.
// Tile t+1 streams into LDS (ASYNCcnt) while tile t runs on the WMMA pipe.
__global__ __launch_bounds__(128) void attn_kernel(
    const __bf16* __restrict__ qT, const __bf16* __restrict__ kT,
    const __bf16* __restrict__ vM, float* __restrict__ out) {
  __shared__ __bf16 Qs[64 * CPAD];        // A: Qs[n][c]
  __shared__ __bf16 KsT[2][64 * CPAD];    // B^T: KsT[m][c], ping-pong
  __shared__ __bf16 Vs[2][128 * MPAD];    // A: Vs[c][m], ping-pong
  __shared__ __bf16 PsT[64 * MPAD];       // B^T: PsT[n][m]
  __shared__ float  rowScale[64];
  __shared__ float  rowSum[64];
  const unsigned qs_o = lds_off32(Qs);
  const unsigned ks_o[2] = { lds_off32(KsT[0]), lds_off32(KsT[1]) };
  const unsigned vs_o[2] = { lds_off32(Vs[0]), lds_off32(Vs[1]) };

  const int tid  = threadIdx.x;
  const int lane = tid & 31;
  const int wv   = tid >> 5;
  const int col  = lane & 15;
  const int hi   = (lane >> 4) << 3;
  const int nbase = blockIdx.x * 64;
  const int b     = blockIdx.y;

  // 16 async b128 copies per K/V tile (8 K rows-of-16 + 8 V rows-of-16 per thread set)
  auto stage_kv = [&](int mt, int buf) {
    const int mbase = mt * 64;
    int r0 = tid >> 4, seg = tid & 15;    // K tile: KsT[m][c], 64 x 256B rows
#pragma unroll
    for (int t = 0; t < 8; ++t) {
      int m = t * 8 + r0;
      async_ld16(ks_o[buf] + m * (CPAD * 2) + seg * 16,
                 kT + (size_t)(b * NPIX + mbase + m) * CDIM + seg * 8);
    }
    int r1 = tid >> 3, sg1 = tid & 7;     // V tile: Vs[c][m], 128 x 128B rows
#pragma unroll
    for (int t = 0; t < 8; ++t) {
      int c = t * 16 + r1;
      async_ld16(vs_o[buf] + c * (MPAD * 2) + sg1 * 16,
                 vM + (size_t)(b * CDIM + c) * NPIX + mbase + sg1 * 8);
    }
  };

  // prologue: Q tile (8 copies) + K/V tile 0
  {
    int r0 = tid >> 4, seg = tid & 15;
#pragma unroll
    for (int t = 0; t < 8; ++t) {
      int n = t * 8 + r0;
      async_ld16(qs_o + n * (CPAD * 2) + seg * 16,
                 qT + (size_t)(b * NPIX + nbase + n) * CDIM + seg * 8);
    }
  }
  stage_kv(0, 0);

  float Mrun[8], Lrun[8];
#pragma unroll
  for (int r = 0; r < 8; ++r) { Mrun[r] = -1e30f; Lrun[r] = 0.f; }
  v8f Oacc[2][4] = {};

  for (int mt = 0; mt < NT; ++mt) {
    const int buf = mt & 1;
    // prefetch next tile into the other buffer, then wait for the current one.
    // async loads complete in order: <=16 outstanding => current tile landed.
    if (mt + 1 < NT) { stage_kv(mt + 1, buf ^ 1); wait_async16(); }
    else             { wait_async0(); }
    __syncthreads();

    // S tile: wave wv owns attention rows [16*wv, 16*wv+16)
    v8f s[4] = {};
#pragma unroll
    for (int kc = 0; kc < 4; ++kc) {
      v16bf a = fragA(Qs + wv * 16 * CPAD + kc * 32, CPAD, lane);
#pragma unroll
      for (int j = 0; j < 4; ++j) {
        v16bf bb = fragBT(KsT[buf] + j * 16 * CPAD + kc * 32, CPAD, lane);
        s[j] = wmma_bf16(a, bb, s[j]);
      }
    }

    // online softmax: row = wv*16 + r + hi spans 16 lanes of a half-wave
    float mnew[8], sc[8], tsum[8];
#pragma unroll
    for (int r = 0; r < 8; ++r) {
      float tm = s[0][r];
#pragma unroll
      for (int j = 1; j < 4; ++j) tm = fmaxf(tm, s[j][r]);
#pragma unroll
      for (int msk = 1; msk < 16; msk <<= 1) tm = fmaxf(tm, __shfl_xor(tm, msk, 32));
      mnew[r] = fmaxf(Mrun[r], tm);
      sc[r]   = __expf(Mrun[r] - mnew[r]);
      tsum[r] = 0.f;
    }
#pragma unroll
    for (int j = 0; j < 4; ++j) {
#pragma unroll
      for (int r = 0; r < 8; ++r) {
        float p = __expf(s[j][r] - mnew[r]);
        tsum[r] += p;
        PsT[(wv * 16 + r + hi) * MPAD + j * 16 + col] = (__bf16)p;  // PsT[n][m]
      }
    }
#pragma unroll
    for (int r = 0; r < 8; ++r) {
#pragma unroll
      for (int msk = 1; msk < 16; msk <<= 1) tsum[r] += __shfl_xor(tsum[r], msk, 32);
      Lrun[r] = Lrun[r] * sc[r] + tsum[r];
      Mrun[r] = mnew[r];
    }
    if (col == 0) {
#pragma unroll
      for (int r = 0; r < 8; ++r) rowScale[wv * 16 + r + hi] = sc[r];
    }
    __syncthreads();

    // rescale O (column n = j*16+col is the attention row) and accumulate V P^T
#pragma unroll
    for (int mi = 0; mi < 2; ++mi) {
#pragma unroll
      for (int j = 0; j < 4; ++j) {
        float scl = rowScale[j * 16 + col];
#pragma unroll
        for (int r = 0; r < 8; ++r) Oacc[mi][j][r] *= scl;
      }
    }
#pragma unroll
    for (int kc = 0; kc < 2; ++kc) {
#pragma unroll
      for (int mi = 0; mi < 2; ++mi) {
        v16bf a = fragA(Vs[buf] + (wv * 32 + mi * 16) * MPAD + kc * 32, MPAD, lane);
#pragma unroll
        for (int j = 0; j < 4; ++j) {
          v16bf bb = fragBT(PsT + j * 16 * MPAD + kc * 32, MPAD, lane);
          Oacc[mi][j] = wmma_bf16(a, bb, Oacc[mi][j]);
        }
      }
    }
    __syncthreads();   // all waves done with buf & PsT before next refill/overwrite
  }

  if (col == 0) {
#pragma unroll
    for (int r = 0; r < 8; ++r) rowSum[wv * 16 + r + hi] = Lrun[r];
  }
  __syncthreads();
#pragma unroll
  for (int mi = 0; mi < 2; ++mi) {
#pragma unroll
    for (int j = 0; j < 4; ++j) {
      float inv = 1.f / rowSum[j * 16 + col];
      int n = nbase + j * 16 + col;
#pragma unroll
      for (int r = 0; r < 8; ++r) {
        int c = wv * 32 + mi * 16 + r + hi;
        out[(size_t)(b * CDIM + c) * NPIX + n] = Oacc[mi][j][r] * inv;
      }
    }
  }
}

extern "C" void kernel_launch(void* const* d_in, const int* in_sizes, int n_in,
                              void* d_out, int out_size, void* d_ws, size_t ws_size,
                              hipStream_t stream) {
  const float* feats  = (const float*)d_in[0];
  const float* proj_w = (const float*)d_in[1];
  const float* proj_b = (const float*)d_in[2];
  const float* w1 = (const float*)d_in[3];
  const float* b1 = (const float*)d_in[4];
  const float* w2 = (const float*)d_in[5];
  const float* b2 = (const float*)d_in[6];
  const float* w3 = (const float*)d_in[7];
  const float* b3 = (const float*)d_in[8];
  // d_in[9] = alpha, unused (reference returns feat_e only)
  float* out = (float*)d_out;

  char* ws = (char*)d_ws;
  const size_t seg = (size_t)BDIM * CDIM * NPIX * 2;  // 4 MB bf16 each
  __bf16* xT = (__bf16*)(ws);
  __bf16* qT = (__bf16*)(ws + seg);
  __bf16* kT = (__bf16*)(ws + 2 * seg);
  __bf16* vM = (__bf16*)(ws + 3 * seg);

  proj_kernel<<<dim3(NPIX / 64, CDIM / 64, BDIM), 128, 0, stream>>>(
      feats, proj_w, proj_b, xT);
  qkv_kernel<<<dim3(NPIX / 64, CDIM / 64, BDIM * 3), 128, 0, stream>>>(
      xT, w1, b1, w2, b2, w3, b3, qT, kT, vM);
  attn_kernel<<<dim3(NPIX / 64, BDIM), 128, 0, stream>>>(qT, kT, vM, out);
}
